// LIF_62423054680210
// MI455X (gfx1250) — matrix operand order repaired and checked by reading, not compiled, not run on hardware
//
#include <hip/hip_runtime.h>

// ---------------------------------------------------------------------------
// SNN (snntorch Leaky) forward, T=100, B=1024, H=256, O=128, on gfx1250.
// Design: 64 persistent workgroups, one per 16-row batch tile. All membrane
// state lives in LDS across the whole time loop. Per-step GEMMs use
// v_wmma_f32_16x16x32_f16 (f16 inputs, f32 accum). Weights are converted to
// f16 and transposed to [K][N] once by a prologue kernel so B fragments are
// 2B per-lane loads, 32B contiguous per k across 16 lanes (hit L2 every step; 192MB L2 holds all 448KB).
// ---------------------------------------------------------------------------

typedef __attribute__((ext_vector_type(16))) _Float16 v16h;
typedef __attribute__((ext_vector_type(8)))  _Float16 v8h;
typedef __attribute__((ext_vector_type(8)))  float    v8f;

#define SHUF16(lo, hi) __builtin_shufflevector((lo), (hi), 0,1,2,3,4,5,6,7,8,9,10,11,12,13,14,15)

__device__ __forceinline__ v8f wmma_f16(v16h a, v16h b, v8f c) {
  // (neg_a, A, neg_b, B, c_mod, C, reuse_a, reuse_b)
  return __builtin_amdgcn_wmma_f32_16x16x32_f16(false, a, false, b, (short)0, c, false, false);
}

__device__ __forceinline__ float clamp01(float v) { return fminf(fmaxf(v, 0.0f), 1.0f); }

// Problem constants
#define T_STEPS 100
#define BATCH   1024
#define HID     256
#define OUTD    128
#define BTILE   16     // batch rows per workgroup
#define NWG     (BATCH / BTILE)   // 64

// d_out layout (flat concat of return tuple, fp32)
#define CUR_OFF 0u
#define MEM_OFF 13107200u              // T*B*O
#define SPK_OFF 13209600u              // + T*B
#define S1_OFF  26316800u              // + T*B*O
#define S2_OFF  26316801u

// workspace layout (halves)
#define W2H_OFF 0
#define W3H_OFF 65536
#define W4H_OFF 131072
#define W5H_OFF 196608
#define WS_HALVES 229376
#define PARTIALS_BYTE_OFF 458752       // WS_HALVES*2, 16B aligned

struct SnnParams {
  const float* x;
  const float* w1; const float* b1;
  const float* b2; const float* b3; const float* b4; const float* b5;
  const float* beta1; const float* thr1;
  const float* beta2; const float* thr2;
  const float* beta3; const float* thr3;
  const float* beta4; const float* thr4;
  const float* beta5; const float* thr5;
  const _Float16* w2h; const _Float16* w3h; const _Float16* w4h; const _Float16* w5h;
  float* cur_rec; float* mem_rec; float* spk_rec; float* partials;
};

// ---------------------------------------------------------------------------
// Prologue: convert fp32 weights -> f16, transposed to [K][N] row-major so
// WMMA B fragments are read 32B contiguous per k across 16 lanes (strided per lane).
// ---------------------------------------------------------------------------
__global__ void snn_convert_weights(const float* __restrict__ w2,
                                    const float* __restrict__ w3,
                                    const float* __restrict__ w4,
                                    const float* __restrict__ w5,
                                    _Float16* __restrict__ out) {
  int i = blockIdx.x * 256 + threadIdx.x;
  if (i < 65536) {                       // w2^T: [256][256]
    int k = i >> 8, n = i & 255;
    out[W2H_OFF + i] = (_Float16)w2[n * 256 + k];
  } else if (i < 131072) {               // w3^T
    int j = i - 65536; int k = j >> 8, n = j & 255;
    out[W3H_OFF + j] = (_Float16)w3[n * 256 + k];
  } else if (i < 196608) {               // w4^T
    int j = i - 131072; int k = j >> 8, n = j & 255;
    out[W4H_OFF + j] = (_Float16)w4[n * 256 + k];
  } else if (i < 229376) {               // w5^T: [256][128]
    int j = i - 196608; int k = j >> 7, n = j & 127;
    out[W5H_OFF + j] = (_Float16)w5[n * 256 + k];
  }
}

// Load A fragment (16x32 f16, M x K) for this lane from an LDS [16][256] tile.
// Lane L: row m=L%16; halves K = c0..c0+7 and c0+16..c0+23, c0 = kc*32 + 8*(L>=16).
__device__ __forceinline__ v16h load_a_frag(const _Float16* act, int lane, int kc) {
  int m = lane & 15;
  int c0 = kc * 32 + ((lane >> 4) << 3);
  const _Float16* p = act + m * 256 + c0;
  v8h lo = *(const v8h*)p;
  v8h hi = *(const v8h*)(p + 16);
  return SHUF16(lo, hi);
}

// Load B fragment (32x16 f16, K x N) from transposed weights wT[K][Nld].
// Lane L holds column N = nt*16 + L%16; halves K = c0..c0+7 and c0+16..c0+23, c0 = kc*32 + 8*(L>=16).
__device__ __forceinline__ v16h load_b_frag(const _Float16* wT, int Nld, int lane, int kc, int nt) {
  int n = nt * 16 + (lane & 15);
  int c0 = kc * 32 + ((lane >> 4) << 3);
  v16h b;
#pragma unroll
  for (int s = 0; s < 8; ++s) {
    b[s] = wT[(size_t)(c0 + s) * Nld + n];
    b[s + 8] = wT[(size_t)(c0 + 16 + s) * Nld + n];
  }
  return b;
}

// Hidden layer: cur = actIn(f16 [16][256]) @ wT + bias; leaky update of
// mstate (LDS f32 [16][256]); stage spike or membrane (f16) into actOut.
template <bool RESET, bool SPIKEOUT>
__device__ __forceinline__ void layer_hidden(
    const _Float16* actIn, _Float16* actOut, const _Float16* __restrict__ wT,
    float* mstate, int lane, int wave,
    float bn0, float be0, float th0,
    float bn1, float be1, float th1,
    float& spkAcc) {
  v16h a[8];
#pragma unroll
  for (int kc = 0; kc < 8; ++kc) a[kc] = load_a_frag(actIn, lane, kc);

#pragma unroll
  for (int pair = 0; pair < 2; ++pair) {
    int nt = wave * 2 + pair;                 // 16 n-tiles over 8 waves
    v8f acc = {};
#pragma unroll
    for (int kc = 0; kc < 8; ++kc) {
      v16h b = load_b_frag(wT, 256, lane, kc, nt);
      acc = wmma_f16(a[kc], b, acc);
    }
    float bn = pair ? bn1 : bn0;
    float be = pair ? be1 : be0;
    float th = pair ? th1 : th0;
    int n  = nt * 16 + (lane & 15);
    int mb = (lane >> 4) << 3;
#pragma unroll
    for (int r = 0; r < 8; ++r) {
      int m = r + mb;                          // C/D layout: M = r + 8*(lane>=16)
      float cur = acc[r] + bn;
      float mo  = mstate[m * 256 + n];
      float rst = (RESET && (mo - th > 0.0f)) ? th : 0.0f;
      float mn  = be * mo + cur - rst;
      mstate[m * 256 + n] = mn;
      if (SPIKEOUT) {
        float s = (mn - th > 0.0f) ? 1.0f : 0.0f;
        actOut[m * 256 + n] = (_Float16)s;
        spkAcc += s;
      } else {
        actOut[m * 256 + n] = (_Float16)mn;
      }
    }
  }
}

// Output layer (O=128): cur5 -> cur_rec; m5 leaky (no reset); s5 -> spk_rec.
__device__ __forceinline__ void layer_out(
    const _Float16* actIn, const _Float16* __restrict__ w5T,
    float* m5state, int lane, int wave,
    float bn, float be, float th,
    float* __restrict__ cur_rec, float* __restrict__ spk_rec,
    int t, int b0) {
  v16h a[8];
#pragma unroll
  for (int kc = 0; kc < 8; ++kc) a[kc] = load_a_frag(actIn, lane, kc);
  int nt = wave;                               // 8 n-tiles over 8 waves
  v8f acc = {};
#pragma unroll
  for (int kc = 0; kc < 8; ++kc) {
    v16h b = load_b_frag(w5T, 128, lane, kc, nt);
    acc = wmma_f16(a[kc], b, acc);
  }
  int n  = nt * 16 + (lane & 15);
  int mb = (lane >> 4) << 3;
#pragma unroll
  for (int r = 0; r < 8; ++r) {
    int m = r + mb;
    float cur = acc[r] + bn;
    float mo  = m5state[m * 128 + n];
    float mn  = be * mo + cur;
    m5state[m * 128 + n] = mn;
    float s = (mn - th > 0.0f) ? 1.0f : 0.0f;
    size_t off = (size_t)t * (BATCH * OUTD) + (size_t)(b0 + m) * OUTD + n;
    cur_rec[off] = cur;
    spk_rec[off] = s;
  }
}

__global__ __launch_bounds__(256) void snn_kernel(SnnParams P) {
  __shared__ alignas(16) float sm1[BTILE * HID];
  __shared__ alignas(16) float sm2[BTILE * HID];
  __shared__ alignas(16) float sm3[BTILE * HID];
  __shared__ alignas(16) float sm4[BTILE * HID];
  __shared__ alignas(16) float sm5[BTILE * OUTD];
  __shared__ alignas(16) _Float16 sactA[BTILE * HID];
  __shared__ alignas(16) _Float16 sactB[BTILE * HID];
  __shared__ float sx[BTILE];
  __shared__ float sred[256];

  const int tid  = threadIdx.x;
  const int lane = tid & 31;
  const int wave = tid >> 5;
  const int b0   = blockIdx.x * BTILE;

  // zero membrane states
  for (int i = tid; i < BTILE * HID; i += 256) {
    sm1[i] = 0.0f; sm2[i] = 0.0f; sm3[i] = 0.0f; sm4[i] = 0.0f;
  }
  for (int i = tid; i < BTILE * OUTD; i += 256) sm5[i] = 0.0f;

  // cache per-thread layer-1 params (h = tid)
  const float w1v = P.w1[tid], b1v = P.b1[tid];
  const float be1 = clamp01(P.beta1[tid]), th1 = P.thr1[tid];

  // cache per-lane epilogue params (n fixed per lane/wave for all t)
  const int nA = (wave * 2) * 16 + (lane & 15);
  const int nB = (wave * 2 + 1) * 16 + (lane & 15);
  const float b2a = P.b2[nA], be2a = clamp01(P.beta2[nA]), th2a = P.thr2[nA];
  const float b2b = P.b2[nB], be2b = clamp01(P.beta2[nB]), th2b = P.thr2[nB];
  const float b3a = P.b3[nA], be3a = clamp01(P.beta3[nA]), th3a = P.thr3[nA];
  const float b3b = P.b3[nB], be3b = clamp01(P.beta3[nB]), th3b = P.thr3[nB];
  const float b4a = P.b4[nA], be4a = clamp01(P.beta4[nA]), th4a = P.thr4[nA];
  const float b4b = P.b4[nB], be4b = clamp01(P.beta4[nB]), th4b = P.thr4[nB];
  const int n5 = wave * 16 + (lane & 15);
  const float b5v = P.b5[n5], be5 = clamp01(P.beta5[n5]), th5 = P.thr5[n5];

  float s1acc = 0.0f, s2acc = 0.0f;
  __syncthreads();

  for (int t = 0; t < T_STEPS; ++t) {
    if (tid < BTILE) sx[tid] = P.x[t * BATCH + b0 + tid];
    __syncthreads();

    // ---- Layer 1 (elementwise: cur = x*w1 + b1), reset=True, spikes -> sactA
    {
      const float* __restrict__ xs = sx;
#pragma unroll 4
      for (int b = 0; b < BTILE; ++b) {
        float mo  = sm1[b * 256 + tid];
        float cur = xs[b] * w1v + b1v;
        float rst = (mo - th1 > 0.0f) ? th1 : 0.0f;
        float mn  = be1 * mo + cur - rst;
        sm1[b * 256 + tid] = mn;
        float s = (mn - th1 > 0.0f) ? 1.0f : 0.0f;
        sactA[b * 256 + tid] = (_Float16)s;
        s1acc += s;
      }
    }
    __syncthreads();

    // ---- Layer 2: s1 @ w2^T, reset=True, spikes -> sactB (accumulate s2)
    layer_hidden<true, true>(sactA, sactB, P.w2h, sm2, lane, wave,
                             b2a, be2a, th2a, b2b, be2b, th2b, s2acc);
    __syncthreads();

    // ---- Layer 3: s2 @ w3^T, reset=True, stage membrane m3 -> sactA
    layer_hidden<true, false>(sactB, sactA, P.w3h, sm3, lane, wave,
                              b3a, be3a, th3a, b3b, be3b, th3b, s2acc);
    __syncthreads();

    // ---- Layer 4: m3 @ w4^T, reset=False, stage membrane m4 -> sactB
    layer_hidden<false, false>(sactA, sactB, P.w4h, sm4, lane, wave,
                               b4a, be4a, th4a, b4b, be4b, th4b, s2acc);
    __syncthreads();

    // ---- Layer 5: m4 @ w5^T, outputs cur_rec/spk_rec, update m5
    layer_out(sactB, P.w5h, sm5, lane, wave, b5v, be5, th5,
              P.cur_rec, P.spk_rec, t, b0);
    __syncthreads();

    // ---- mem_rec[t, b] = mean(m5[b, :])
    if (tid < BTILE) {
      float s = 0.0f;
      for (int n = 0; n < OUTD; ++n) s += sm5[tid * OUTD + n];
      P.mem_rec[(size_t)t * BATCH + b0 + tid] = s * (1.0f / OUTD);
    }
    // loop-top barrier (after sx write) protects reduction vs next step
  }

  // ---- deterministic per-WG spike sums -> workspace partials
  __syncthreads();
  sred[tid] = s1acc;
  __syncthreads();
  if (tid == 0) {
    float a = 0.0f;
    for (int i = 0; i < 256; ++i) a += sred[i];
    P.partials[blockIdx.x * 2] = a;
  }
  __syncthreads();
  sred[tid] = s2acc;
  __syncthreads();
  if (tid == 0) {
    float a = 0.0f;
    for (int i = 0; i < 256; ++i) a += sred[i];
    P.partials[blockIdx.x * 2 + 1] = a;
  }
}

__global__ void snn_finalize(const float* __restrict__ partials,
                             float* __restrict__ s1o, float* __restrict__ s2o) {
  float a = 0.0f, b = 0.0f;
  for (int i = 0; i < NWG; ++i) { a += partials[2 * i]; b += partials[2 * i + 1]; }
  const float inv = 1.0f / (float)(T_STEPS * BATCH * HID);  // 26,214,400
  *s1o = a * inv;
  *s2o = b * inv;
}

extern "C" void kernel_launch(void* const* d_in, const int* in_sizes, int n_in,
                              void* d_out, int out_size, void* d_ws, size_t ws_size,
                              hipStream_t stream) {
  // setup_inputs() order: x, then params dict insertion order.
  const float* x     = (const float*)d_in[0];
  const float* w1    = (const float*)d_in[1];
  const float* b1    = (const float*)d_in[2];
  const float* w2    = (const float*)d_in[3];
  const float* b2    = (const float*)d_in[4];
  const float* w3    = (const float*)d_in[5];
  const float* b3    = (const float*)d_in[6];
  const float* w4    = (const float*)d_in[7];
  const float* b4    = (const float*)d_in[8];
  const float* w5    = (const float*)d_in[9];
  const float* b5    = (const float*)d_in[10];
  const float* beta1 = (const float*)d_in[11];
  const float* thr1  = (const float*)d_in[12];
  const float* beta2 = (const float*)d_in[13];
  const float* thr2  = (const float*)d_in[14];
  const float* beta3 = (const float*)d_in[15];
  const float* thr3  = (const float*)d_in[16];
  const float* beta4 = (const float*)d_in[17];
  const float* thr4  = (const float*)d_in[18];
  const float* beta5 = (const float*)d_in[19];
  const float* thr5  = (const float*)d_in[20];

  float* out = (float*)d_out;
  _Float16* wsh = (_Float16*)d_ws;
  float* partials = (float*)((char*)d_ws + PARTIALS_BYTE_OFF);

  // 1) one-time weight convert/transpose to f16 [K][N]
  snn_convert_weights<<<(WS_HALVES + 255) / 256, 256, 0, stream>>>(w2, w3, w4, w5, wsh);

  // 2) persistent SNN kernel: 64 WGs x 256 threads (8 wave32s)
  SnnParams P;
  P.x = x; P.w1 = w1; P.b1 = b1; P.b2 = b2; P.b3 = b3; P.b4 = b4; P.b5 = b5;
  P.beta1 = beta1; P.thr1 = thr1; P.beta2 = beta2; P.thr2 = thr2;
  P.beta3 = beta3; P.thr3 = thr3; P.beta4 = beta4; P.thr4 = thr4;
  P.beta5 = beta5; P.thr5 = thr5;
  P.w2h = wsh + W2H_OFF; P.w3h = wsh + W3H_OFF; P.w4h = wsh + W4H_OFF; P.w5h = wsh + W5H_OFF;
  P.cur_rec = out + CUR_OFF; P.mem_rec = out + MEM_OFF; P.spk_rec = out + SPK_OFF;
  P.partials = partials;
  snn_kernel<<<NWG, 256, 0, stream>>>(P);

  // 3) deterministic scalar reduction for mean(s1), mean(s2)
  snn_finalize<<<1, 1, 0, stream>>>(partials, out + S1_OFF, out + S2_OFF);
}
